// PLPConv_3221225472212
// MI455X (gfx1250) — matrix operand order repaired and checked by reading, not compiled, not run on hardware
//
#include <hip/hip_runtime.h>

// ---------------------------------------------------------------------------
// GAT-style propagation for MI455X (gfx1250, wave32).
//   N=100000 nodes, E=3200000 edges, C=64 channels.
//   out = [ rst (N*C f32) | a (E f32) ]
// L2-resident / atomic-bound problem (192MB L2 holds soft_label + rst).
// v_wmma_f32_16x16x32_f16 performs diag(a) x F per 16-edge x 16-ch tile.
// ---------------------------------------------------------------------------

typedef __attribute__((ext_vector_type(16))) _Float16 v16h;
typedef __attribute__((ext_vector_type(8)))  float    v8f;

#define NN 100000
#define NE 3200000
#define CC 64

// Order-preserving int encoding of float for atomicMax-based segment-max.
__device__ __forceinline__ int f32_key(float f) {
    int i = __float_as_int(f);
    return (i >= 0) ? i : (i ^ 0x7FFFFFFF);
}
__device__ __forceinline__ float key_f32(int i) {
    return __int_as_float((i >= 0) ? i : (i ^ 0x7FFFFFFF));
}

// Kernel 1: zero rst, init mkeys = INT_MIN (encodes "-inf"), denom = 0.
__global__ void k_init(float* __restrict__ rst, int* __restrict__ mkeys,
                       float* __restrict__ denom) {
    const long stride = (long)gridDim.x * blockDim.x;
    for (long i = (long)blockIdx.x * blockDim.x + threadIdx.x;
         i < (long)NN * CC; i += stride) {
        rst[i] = 0.0f;
        if (i < NN) { mkeys[i] = INT_MIN; denom[i] = 0.0f; }
    }
}

// Kernel 1b (f16 path): one-time f32 -> f16 copy of soft_label into ws.
__global__ void k_prep(const float* __restrict__ sl, _Float16* __restrict__ slh) {
    const long stride = (long)gridDim.x * blockDim.x;
    for (long i = (long)blockIdx.x * blockDim.x + threadIdx.x;
         i < (long)NN * CC; i += stride)
        slh[i] = (_Float16)sl[i];
}

// Kernel 2: segment-max of per-edge logits over dst.
__global__ void k_segmax(const float* __restrict__ e, const int* __restrict__ dst,
                         int* __restrict__ mkeys) {
    int i = blockIdx.x * blockDim.x + threadIdx.x;
    if (i < NE) atomicMax(&mkeys[dst[i]], f32_key(e[i]));
}

// Kernel 3: ex = exp(e - m[dst]) -> a_out (temporarily); denom[dst] += ex.
__global__ void k_exp(const float* __restrict__ e, const int* __restrict__ dst,
                      const int* __restrict__ mkeys, float* __restrict__ denom,
                      float* __restrict__ a_out) {
    int i = blockIdx.x * blockDim.x + threadIdx.x;
    if (i < NE) {
        int d = dst[i];
        float ex = expf(e[i] - key_f32(mkeys[d]));
        a_out[i] = ex;
        atomicAdd(&denom[d], ex);
    }
}

// Kernel 4: one wave per 16-edge tile.
//   a = ex / denom[dst]  (written back to a_out in place, deterministic)
//   For each 16-channel group g: D(16x16) = diag(a) x F_tile via
//   v_wmma_f32_16x16x32_f16, then non-returning atomic scatter into rst.
//
// WMMA operand layouts (wave32):
//   A (16x32 f16): lanes 0-15 elem k -> K=k; lanes 16-31 elem k -> K=16+k.
//     diag(a) lives in K=0..15; hi-lane elements (K=16..31) are zero.
//   B (32x16 f16): lanes 0-15 elem k -> K=k (feature rows), col = lane%16.
//     Hi lanes (K=16..31) carry duplicated finite values — they multiply the
//     structurally-zero hi-lane A elements, so NO per-element zero-select needed.
//   C/D (16x16 f32): VGPR r: lanes 0-15 -> M=r, lanes 16-31 -> M=8+r; N=lane%16.
//
// Cross-lane data movement: dst rows via 8 hoisted ds_bpermute (__shfl);
// src rows via readlane -> uniform SGPR row pointers (saddr-form u16 loads).
template <bool USE_F16>
__global__ void __launch_bounds__(256)
k_scatter(const float* __restrict__ sl, const _Float16* __restrict__ slh,
          const int* __restrict__ src, const int* __restrict__ dst,
          const float* __restrict__ denom, float* __restrict__ rst,
          float* __restrict__ a_out) {
    const int lane = threadIdx.x & 31;
    const int tile = blockIdx.x * (blockDim.x >> 5) + (threadIdx.x >> 5);
    const int base = tile * 16;
    if (base >= NE) return;                 // uniform across the wave

    const int  m  = lane & 15;              // edge-in-tile / column index
    const bool hi = lane >= 16;
    const int  ed = base + m;

    const int   s_v = src[ed];
    const int   d_v = dst[ed];
    const float ex  = a_out[ed];
    const float a   = ex / denom[d_v];      // edge-softmax value
    if (!hi) a_out[ed] = a;                 // final attention output

    // Per-lane scatter row offsets: lane needs dst of rows r (lo) / r+8 (hi).
    // 8 ds_bpermute ops, hoisted; constant-indexed VGPR array.
    const int off8 = (lane & 16) >> 1;      // 0 (lo half) | 8 (hi half)
    int rbase[8];
#pragma unroll
    for (int r = 0; r < 8; ++r) {
        const int dd = __shfl(d_v, r + off8, 32);
        rbase[r] = dd * CC + m;             // + g*16 added per group
    }

    // Uniform row pointers (readlane -> SGPR pair per row): encourages
    // saddr-form global_load_u16 with the shared ch offset in one VGPR.
    const _Float16* rowh[16];
    const float*    rowf[16];
#pragma unroll
    for (int k = 0; k < 16; ++k) {
        const int s = __builtin_amdgcn_readlane(s_v, k);
        if (USE_F16) rowh[k] = slh + (size_t)s * CC;
        else         rowf[k] = sl  + (size_t)s * CC;
    }

    // A = diag(a) embedded in K=0..15 of a 16x32 f16 matrix (rest zero).
    v16h A = {};
#pragma unroll
    for (int k = 0; k < 16; ++k) {
        const int K = hi ? (16 + k) : k;
        A[k] = (K == m) ? (_Float16)a : (_Float16)0.0f;
    }

#pragma unroll
    for (int g = 0; g < 4; ++g) {
        const int ch = g * 16 + m;          // this lane's channel/column
        // B rows K=0..15 = gathered feature rows; hi lanes duplicate lo-lane
        // addresses (finite data x zero A elements), no zero-select.
        v16h B;
#pragma unroll
        for (int k = 0; k < 16; ++k)
            B[k] = USE_F16 ? rowh[k][ch] : (_Float16)rowf[k][ch];

        v8f c = {};
        c = __builtin_amdgcn_wmma_f32_16x16x32_f16(
                /*neg_a=*/false, A, /*neg_b=*/false, B,
                /*c_mod=*/(short)0, c, /*reuse_a=*/false, /*reuse_b=*/false);
        // Non-returning f32 atomics -> global_atomic_add_f32 (STOREcnt).
#pragma unroll
        for (int r = 0; r < 8; ++r)
            atomicAdd(&rst[(size_t)(rbase[r] + g * 16)], c[r]);
    }
}

extern "C" void kernel_launch(void* const* d_in, const int* in_sizes, int n_in,
                              void* d_out, int out_size, void* d_ws, size_t ws_size,
                              hipStream_t stream) {
    // inputs: [0]=i (unused scalar), [1]=src i32[E], [2]=dst i32[E],
    //         [3]=e f32[E], [4]=soft_label f32[N*C]
    const int*   src = (const int*)d_in[1];
    const int*   dst = (const int*)d_in[2];
    const float* e   = (const float*)d_in[3];
    const float* sl  = (const float*)d_in[4];

    float* rst   = (float*)d_out;                       // N*C
    float* a_out = rst + (size_t)NN * CC;               // E

    // workspace: mkeys int[N] | denom f32[N] | slh f16[N*C]
    int*      mkeys = (int*)d_ws;
    float*    denom = (float*)((char*)d_ws + (size_t)NN * sizeof(int));
    _Float16* slh   = (_Float16*)((char*)d_ws + (size_t)NN * 8);
    const size_t need_f16 = (size_t)NN * 8 + (size_t)NN * CC * sizeof(_Float16);
    const bool use_f16 = (ws_size >= need_f16);

    const int T = 256;
    k_init   <<<(NN * CC + T - 1) / T, T, 0, stream>>>(rst, mkeys, denom);
    if (use_f16)
        k_prep<<<(NN * CC + T - 1) / T, T, 0, stream>>>(sl, slh);
    k_segmax <<<(NE + T - 1) / T, T, 0, stream>>>(e, dst, mkeys);
    k_exp    <<<(NE + T - 1) / T, T, 0, stream>>>(e, dst, mkeys, denom, a_out);

    // one wave (32 lanes) per 16-edge tile; 8 waves per block.
    const int tiles  = NE / 16;                         // 200000 (exact)
    const int blocks = (tiles + 7) / 8;                 // 25000
    if (use_f16)
        k_scatter<true><<<blocks, T, 0, stream>>>(sl, slh, src, dst, denom, rst, a_out);
    else
        k_scatter<false><<<blocks, T, 0, stream>>>(sl, slh, src, dst, denom, rst, a_out);
}
